// GCN_73100343378654
// MI455X (gfx1250) — compile-verified, matrix operand order checked by reading, not run on hardware
//
#include <hip/hip_runtime.h>
#include <hip/hip_bf16.h>

// ---------------- constants from the reference ----------------
#define DIN   128
#define HID   96
#define HEADS 4
#define FH    24
#define GNUM  128
#define OUTF  128
#define FDIM  (HID + DIN)   // 224
#define KMAX  224           // largest GEMM K (FDIM)
#define SLOPE 0.2f
#define EPS   1e-5f

typedef __attribute__((ext_vector_type(16))) _Float16 v16h;
typedef __attribute__((ext_vector_type(8)))  float    v8f;
typedef __attribute__((ext_vector_type(4)))  int      v4i;

// address-space-qualified v4i pointers for async global->LDS copies
// (clang's gfx1250 builtin takes v4i AS(1)* / v4i AS(3)* per the diagnostic)
typedef v4i __attribute__((address_space(1)))* gv4p;
typedef v4i __attribute__((address_space(3)))* lv4p;

#if defined(__gfx1250__) && __has_builtin(__builtin_amdgcn_global_load_async_to_lds_b128)
#define HAS_ASYNC_COPY 1
#else
#define HAS_ASYNC_COPY 0
#endif

__device__ __forceinline__ void wait_async_zero() {
#if __has_builtin(__builtin_amdgcn_s_wait_asynccnt)
    __builtin_amdgcn_s_wait_asynccnt(0);
#elif defined(__gfx1250__)
    asm volatile("s_wait_asynccnt 0x0" ::: "memory");
#endif
}

// ---------------- small device helpers ----------------
__device__ __forceinline__ float leaky(float v) { return v >= 0.f ? v : SLOPE * v; }

// order-preserving float<->uint encoding for atomicMax-based segment max
__device__ __forceinline__ unsigned fenc(float f) {
    unsigned u = __float_as_uint(f);
    return (u >> 31) ? ~u : (u | 0x80000000u);
}
__device__ __forceinline__ float fdec(unsigned u) {
    return (u >> 31) ? __uint_as_float(u & 0x7FFFFFFFu) : __uint_as_float(~u);
}

// edge e in [0, E+N): first E are edge_index columns, last N are self-loops
__device__ __forceinline__ void edge_sd(const int* ei, int E, int e, int& s, int& d) {
    if (e < E) { s = ei[e]; d = ei[E + e]; } else { s = d = e - E; }
}

// wave32 butterfly reduce
__device__ __forceinline__ float wred(float v) {
#pragma unroll
    for (int m = 16; m > 0; m >>= 1) v += __shfl_xor(v, m, 32);
    return v;
}

// ---------------- utility fills ----------------
__global__ void fill_f32(float* p, float v, int n) {
    int i = blockIdx.x * blockDim.x + threadIdx.x;
    if (i < n) p[i] = v;
}
__global__ void fill_u32(unsigned* p, unsigned v, int n) {
    int i = blockIdx.x * blockDim.x + threadIdx.x;
    if (i < n) p[i] = v;
}

// ---------------- LayerNorm over DIN=128, one wave per row ----------------
__global__ void ln_kernel(const float* __restrict__ x, const float* __restrict__ g,
                          const float* __restrict__ b, float* __restrict__ y, int N) {
    int wid = threadIdx.x >> 5, lane = threadIdx.x & 31;
    int row = blockIdx.x * (blockDim.x >> 5) + wid;
    if (row >= N) return;
    const float* xr = x + (size_t)row * DIN;
    float v[4];
    float s = 0.f;
#pragma unroll
    for (int j = 0; j < 4; ++j) { v[j] = xr[lane + 32 * j]; s += v[j]; }
    s = wred(s);
    float mu = s * (1.f / DIN);
    float q = 0.f;
#pragma unroll
    for (int j = 0; j < 4; ++j) { float d = v[j] - mu; q += d * d; }
    q = wred(q);
    float inv = rsqrtf(q * (1.f / DIN) + EPS);
    float* yr = y + (size_t)row * DIN;
#pragma unroll
    for (int j = 0; j < 4; ++j) {
        int c = lane + 32 * j;
        yr[c] = g[c] * (v[j] - mu) * inv + b[c];
    }
}

// ---------------- weight pre-pack: f32 [K,Nc] -> f16 in WMMA B-lane layout ----
// w16 element order: [ks][tn][lane][i], i.e. for k-step ks and col-tile tn, lane
// reads its 16 halves contiguously (one 32B load per wmma).
__global__ void pack_w16(const float* __restrict__ W, _Float16* __restrict__ w16,
                         int K, int Nc) {
    int idx = blockIdx.x * blockDim.x + threadIdx.x;
    if (idx >= K * Nc) return;
    int ntile = Nc >> 4;
    int i    = idx & 15;
    int ln   = (idx >> 4) & 31;
    int rest = idx >> 9;
    int tn = rest % ntile, ks = rest / ntile;
    int col = (tn << 4) + (ln & 15);
    int kg  = ln >> 4;
    int k   = (ks << 5) + 8 * kg + (i & 7) + ((i >> 3) << 4);
    w16[idx] = (_Float16)W[(size_t)k * Nc + col];
}

// ---------------- WMMA GEMM: C[M,Nc] = act(A[M,K] @ W[K,Nc] + bias) ----------------
// Requires K%32==0, Nc%16==0, M%16==0, Nc/16 <= 8 (true for all call sites).
// One block per 16-row slab of A. The slab is staged into LDS (async global->LDS
// when available), converted once to f16 in the ISA 7.12.2 A-lane layout, then
// each wave computes one 16x16 output tile per col-tile with
// v_wmma_f32_16x16x32_f16 (f32 accumulate), reading B from the pre-packed
// global f16 buffer (one contiguous 32B load per lane per k-step).
__global__ __launch_bounds__(256) void gemm_wmma(
    const float* __restrict__ A, const _Float16* __restrict__ w16,
    const float* __restrict__ bias, float* __restrict__ C,
    int M, int K, int Nc, int act) {
    __shared__ __align__(16) float    sA [16 * KMAX];
    __shared__ __align__(32) _Float16 sAh[16 * KMAX];
    const int tid  = threadIdx.x;
    const int lane = tid & 31, wid = tid >> 5;
    const int ntile = Nc >> 4;
    const int row0  = blockIdx.x << 4;
    const int slab  = 16 * K;                      // floats (== halves in sAh)

    // keep the shared weight buffer hot (reused by every block)
    for (int off = tid * 64; off < K * Nc * 2; off += 256 * 64)
        __builtin_prefetch((const char*)w16 + off, 0, 2);

    // ---- stage A slab (16 rows x K f32, contiguous) into LDS ----
    const float* gsrc = A + (size_t)row0 * K;
#if HAS_ASYNC_COPY
    for (int c = tid; c * 4 < slab; c += 256) {
        __builtin_amdgcn_global_load_async_to_lds_b128(
            (gv4p)(gsrc + c * 4), (lv4p)(&sA[c * 4]), 0, 0);
    }
    wait_async_zero();
#else
    for (int c = tid; c * 4 < slab; c += 256) {
        *(float4*)&sA[c * 4] = *(const float4*)(gsrc + c * 4);
    }
#endif
    __syncthreads();

    // ---- rearrange to f16 WMMA A-lane layout: [ks][lane][i] ----
    for (int idx = tid; idx < slab; idx += 256) {
        int i  = idx & 15;
        int ln = (idx >> 4) & 31;
        int ks = idx >> 9;
        int r  = ln & 15, kg = ln >> 4;
        int k  = (ks << 5) + 8 * kg + (i & 7) + ((i >> 3) << 4);
        sAh[idx] = (_Float16)sA[r * K + k];
    }
    __syncthreads();

    // ---- compute: wave w handles col tile w (wave-uniform predicate) ----
    if (wid < ntile) {
        v8f acc = {};
        const int ksteps = K >> 5;
        for (int ks = 0; ks < ksteps; ++ks) {
            v16h a = *(const v16h*)&sAh[((ks << 5) + lane) << 4];
            v16h b = *(const v16h*)&w16[(size_t)(((ks * ntile + wid) << 5) + lane) << 4];
            acc = __builtin_amdgcn_wmma_f32_16x16x32_f16(
                false, a, false, b, (short)0, acc, false, false);
        }
        const int r = lane & 15, kg = lane >> 4;
#pragma unroll
        for (int i = 0; i < 8; ++i) {
            int m = row0 + i + 8 * kg;
            int n = (wid << 4) + r;
            if (m >= M) continue;
            float v = acc[i];
            if (bias) v += bias[n];
            if (act == 1) v = leaky(v);
            else if (act == 2) v = tanhf(v);
            C[(size_t)m * Nc + n] = v;
        }
    }
}

// ---------------- GCN: degree + normalized scatter ----------------
__global__ void deg_edges(const int* __restrict__ ei, int E, float* __restrict__ deg) {
    int e = blockIdx.x * blockDim.x + threadIdx.x;
    if (e < E) atomicAdd(&deg[ei[E + e]], 1.0f);
}
__global__ void deg_fin(float* dinv, int N) {
    int i = blockIdx.x * blockDim.x + threadIdx.x;
    if (i < N) dinv[i] = rsqrtf(dinv[i]);      // deg >= 1 (self-loop)
}
__global__ void gcn_scatter(const int* __restrict__ ei, int E, int EN,
                            const float* __restrict__ dinv,
                            const float* __restrict__ hh, float* __restrict__ out) {
    int t = blockIdx.x * blockDim.x + threadIdx.x;
    int e = t >> 5, lane = t & 31;
    if (e >= EN) return;
    int s, d; edge_sd(ei, E, e, s, d);
    const float* hs = hh + (size_t)s * HID;
    __builtin_prefetch(&hs[lane], 0, 0);       // pull src row lines early
    float nrm = dinv[s] * dinv[d];
    float* od = out + (size_t)d * HID;
#pragma unroll
    for (int j = 0; j < 3; ++j) {
        int f = lane + 32 * j;
        atomicAdd(&od[f], hs[f] * nrm);
    }
}

// ---------------- node-level BatchNorm (conv bias cancels in (x-mu)) ----------------
__global__ void bn_stats(const float* __restrict__ x, float* __restrict__ sum,
                         float* __restrict__ sq, int N) {
    int c = threadIdx.x;                       // 96 threads
    int r0 = blockIdx.x * 64;
    int r1 = (r0 + 64 < N) ? r0 + 64 : N;
    float s = 0.f, q = 0.f;
    for (int r = r0; r < r1; ++r) {
        float v = x[(size_t)r * HID + c];
        s += v; q += v * v;
    }
    atomicAdd(&sum[c], s);
    atomicAdd(&sq[c], q);
}
__global__ void bn_final(const float* sum, const float* sq, const float* g,
                         const float* b, float* scale, float* shift, int N) {
    int c = threadIdx.x;
    float invn = 1.f / (float)N;
    float mu = sum[c] * invn;
    float var = sq[c] * invn - mu * mu;
    float sc = g[c] * rsqrtf(var + EPS);
    scale[c] = sc;
    shift[c] = b[c] - mu * sc;
}
__global__ void bn_apply_leaky(float* __restrict__ x, const float* __restrict__ scale,
                               const float* __restrict__ shift, int n) {
    int i = blockIdx.x * blockDim.x + threadIdx.x;
    if (i >= n) return;
    int c = i % HID;
    x[i] = leaky(x[i] * scale[c] + shift[c]);
}

// ---------------- GAT attention ----------------
__global__ void gat_coef(const float* __restrict__ hh, const float* __restrict__ asrc,
                         const float* __restrict__ adst, float* __restrict__ as_,
                         float* __restrict__ ad_, int N) {
    int idx = blockIdx.x * blockDim.x + threadIdx.x;
    if (idx >= N * HEADS) return;
    int n = idx >> 2, hd = idx & 3;
    const float* row = hh + (size_t)n * HID + hd * FH;
    const float* ar = asrc + hd * FH;
    const float* ad = adst + hd * FH;
    float s = 0.f, d = 0.f;
#pragma unroll
    for (int f = 0; f < FH; ++f) { s += row[f] * ar[f]; d += row[f] * ad[f]; }
    as_[idx] = s; ad_[idx] = d;
}
__global__ void gat_max(const int* __restrict__ ei, int E, int EN,
                        const float* __restrict__ as_, const float* __restrict__ ad_,
                        unsigned* __restrict__ menc) {
    int idx = blockIdx.x * blockDim.x + threadIdx.x;
    if (idx >= EN * HEADS) return;
    int e = idx >> 2, hd = idx & 3;
    int s, d; edge_sd(ei, E, e, s, d);
    float v = leaky(as_[s * HEADS + hd] + ad_[d * HEADS + hd]);
    atomicMax(&menc[d * HEADS + hd], fenc(v));
}
__global__ void gat_expsum(const int* __restrict__ ei, int E, int EN,
                           const float* __restrict__ as_, const float* __restrict__ ad_,
                           const unsigned* __restrict__ menc, float* __restrict__ alp,
                           float* __restrict__ ssum) {
    int idx = blockIdx.x * blockDim.x + threadIdx.x;
    if (idx >= EN * HEADS) return;
    int e = idx >> 2, hd = idx & 3;
    int s, d; edge_sd(ei, E, e, s, d);
    float v = leaky(as_[s * HEADS + hd] + ad_[d * HEADS + hd]);
    float ex = __expf(v - fdec(menc[d * HEADS + hd]));
    alp[idx] = ex;
    atomicAdd(&ssum[d * HEADS + hd], ex);
}
__global__ void gat_scatter(const int* __restrict__ ei, int E, int EN,
                            const float* __restrict__ hh, const float* __restrict__ alp,
                            const float* __restrict__ ssum, float* __restrict__ out) {
    int t = blockIdx.x * blockDim.x + threadIdx.x;
    int e = t >> 5, lane = t & 31;
    if (e >= EN) return;
    int s, d; edge_sd(ei, E, e, s, d);
    const float* hs = hh + (size_t)s * HID;
    __builtin_prefetch(&hs[lane], 0, 0);
    float* od = out + (size_t)d * HID;
#pragma unroll
    for (int j = 0; j < 3; ++j) {
        int f = lane + 32 * j;
        int hd = f / FH;
        float w = alp[e * HEADS + hd] / ssum[d * HEADS + hd];
        atomicAdd(&od[f], hs[f] * w);
    }
}

// ---------------- GlobalAttention pooling ----------------
__global__ void gate_dot(const float* __restrict__ g1, const float* __restrict__ w2,
                         float* __restrict__ gate, int N) {
    int t = blockIdx.x * blockDim.x + threadIdx.x;
    int n = t >> 5, lane = t & 31;
    if (n >= N) return;
    const float* r = g1 + (size_t)n * HID;
    float s = 0.f;
#pragma unroll
    for (int j = 0; j < 3; ++j) { int f = lane + 32 * j; s += r[f] * w2[f]; }
    s = wred(s);
    if (lane == 0) gate[n] = s;                // gate_b2 cancels in softmax
}
__global__ void pool_max(const int* __restrict__ batch, const float* __restrict__ gate,
                         unsigned* __restrict__ m2, int N) {
    int n = blockIdx.x * blockDim.x + threadIdx.x;
    if (n < N) atomicMax(&m2[batch[n]], fenc(gate[n]));
}
__global__ void pool_expsum(const int* __restrict__ batch, float* __restrict__ gate,
                            const unsigned* __restrict__ m2, float* __restrict__ psum, int N) {
    int n = blockIdx.x * blockDim.x + threadIdx.x;
    if (n >= N) return;
    int g = batch[n];
    float a = __expf(gate[n] - fdec(m2[g]));
    gate[n] = a;
    atomicAdd(&psum[g], a);
}
__global__ void pool_scatter(const int* __restrict__ batch, const float* __restrict__ gate,
                             const float* __restrict__ psum, const float* __restrict__ h,
                             const float* __restrict__ xres, float* __restrict__ gemb,
                             float* __restrict__ xsum, float* __restrict__ cnt, int N) {
    int t = blockIdx.x * blockDim.x + threadIdx.x;
    int n = t >> 5, lane = t & 31;
    if (n >= N) return;
    int g = batch[n];
    float att = gate[n] / psum[g];
    const float* hr = h + (size_t)n * HID;
    const float* xr = xres + (size_t)n * DIN;
#pragma unroll
    for (int j = 0; j < 3; ++j) { int f = lane + 32 * j; atomicAdd(&gemb[g * HID + f], att * hr[f]); }
#pragma unroll
    for (int j = 0; j < 4; ++j) { int f = lane + 32 * j; atomicAdd(&xsum[g * DIN + f], xr[f]); }
    if (lane == 0) atomicAdd(&cnt[g], 1.0f);
}
__global__ void build_f(const float* __restrict__ gemb, const float* __restrict__ xsum,
                        const float* __restrict__ cnt, float* __restrict__ fbuf) {
    int i = blockIdx.x * blockDim.x + threadIdx.x;
    if (i >= GNUM * FDIM) return;
    int g = i / FDIM, c = i % FDIM;
    fbuf[i] = (c < HID) ? gemb[g * HID + c]
                        : xsum[g * DIN + (c - HID)] / fmaxf(cnt[g], 1.0f);
}
// final BatchNorm over the G=128 graphs (out_b cancels)
__global__ void final_bn(const float* __restrict__ o, const float* __restrict__ g,
                         const float* __restrict__ b, float* __restrict__ out) {
    int c = threadIdx.x;                        // 128 threads
    float s = 0.f, q = 0.f;
    for (int r = 0; r < GNUM; ++r) { float v = o[r * OUTF + c]; s += v; q += v * v; }
    float mu = s * (1.f / GNUM);
    float var = q * (1.f / GNUM) - mu * mu;
    float inv = rsqrtf(var + EPS);
    for (int r = 0; r < GNUM; ++r)
        out[r * OUTF + c] = g[c] * (o[r * OUTF + c] - mu) * inv + b[c];
}

// ---------------- host orchestration ----------------
extern "C" void kernel_launch(void* const* d_in, const int* in_sizes, int n_in,
                              void* d_out, int out_size, void* d_ws, size_t ws_size,
                              hipStream_t stream) {
    const float* x     = (const float*)d_in[0];
    const int*   ei    = (const int*)d_in[1];
    const int*   batch = (const int*)d_in[2];
    int pi = 3;
    const float* ln_g   = (const float*)d_in[pi++];
    const float* ln_b   = (const float*)d_in[pi++];
    const float* proj_W = (const float*)d_in[pi++];
    const float* proj_b = (const float*)d_in[pi++];
    const float* W0     = (const float*)d_in[pi++];
    const float* b0     = (const float*)d_in[pi++]; (void)b0;   // cancels in BN
    const float* W1     = (const float*)d_in[pi++];
    const float* b1     = (const float*)d_in[pi++]; (void)b1;
    const float* asrc1  = (const float*)d_in[pi++];
    const float* adst1  = (const float*)d_in[pi++];
    const float* W2     = (const float*)d_in[pi++];
    const float* b2     = (const float*)d_in[pi++]; (void)b2;
    const float* W3     = (const float*)d_in[pi++];
    const float* b3     = (const float*)d_in[pi++]; (void)b3;
    const float* asrc3  = (const float*)d_in[pi++];
    const float* adst3  = (const float*)d_in[pi++];
    const float* bng0   = (const float*)d_in[pi++];
    const float* bnb0   = (const float*)d_in[pi++];
    const float* bng1   = (const float*)d_in[pi++];
    const float* bnb1   = (const float*)d_in[pi++];
    const float* bng2   = (const float*)d_in[pi++];
    const float* bnb2   = (const float*)d_in[pi++];
    const float* bng3   = (const float*)d_in[pi++];
    const float* bnb3   = (const float*)d_in[pi++];
    const float* gateW1 = (const float*)d_in[pi++];
    const float* gateb1 = (const float*)d_in[pi++];
    const float* gateW2 = (const float*)d_in[pi++];
    const float* gateb2 = (const float*)d_in[pi++]; (void)gateb2; // cancels in softmax
    const float* out_W  = (const float*)d_in[pi++];
    const float* out_b  = (const float*)d_in[pi++]; (void)out_b;  // cancels in BN
    const float* obn_g  = (const float*)d_in[pi++];
    const float* obn_b  = (const float*)d_in[pi++];

    const int N  = in_sizes[0] / DIN;
    const int E  = in_sizes[1] / 2;
    const int EN = E + N;
    (void)n_in; (void)out_size; (void)ws_size;

    // workspace carve-out (32B-aligned slices)
    float* ws = (float*)d_ws;
    size_t off = 0;
    auto carve = [&](size_t n) {
        float* p = ws + off;
        off += (n + 7) & ~(size_t)7;
        return p;
    };
    float*    x_res = carve((size_t)N * DIN);
    float*    hb0   = carve((size_t)N * HID);
    float*    hb1   = carve((size_t)N * HID);
    float*    hb2   = carve((size_t)N * HID);
    float*    dinv  = carve(N);
    float*    as_   = carve((size_t)N * HEADS);
    float*    ad_   = carve((size_t)N * HEADS);
    unsigned* menc  = (unsigned*)carve((size_t)N * HEADS);
    float*    ssum  = carve((size_t)N * HEADS);
    float*    alp   = carve((size_t)EN * HEADS);
    float*    gate  = carve(N);
    float*    bnsum = carve(96);
    float*    bnsq  = carve(96);
    float*    bnscl = carve(96);
    float*    bnsh  = carve(96);
    unsigned* m2    = (unsigned*)carve(GNUM);
    float*    psum  = carve(2 * GNUM);
    float*    cnt   = psum + GNUM;               // contiguous with psum
    float*    gemb  = carve((size_t)GNUM * HID);
    float*    xsum  = carve((size_t)GNUM * DIN);
    float*    fbuf  = carve((size_t)GNUM * FDIM);
    float*    obuf  = carve((size_t)GNUM * OUTF);
    _Float16* w16   = (_Float16*)carve(KMAX * OUTF / 2);   // packed-weight scratch

    auto fill = [&](float* p, float v, int n) {
        fill_f32<<<(n + 255) / 256, 256, 0, stream>>>(p, v, n);
    };
    auto fillu = [&](unsigned* p, unsigned v, int n) {
        fill_u32<<<(n + 255) / 256, 256, 0, stream>>>(p, v, n);
    };
    auto gemm = [&](const float* A, const float* W, const float* bias, float* C,
                    int M, int K, int Nc, int act) {
        pack_w16<<<(K * Nc + 255) / 256, 256, 0, stream>>>(W, w16, K, Nc);
        gemm_wmma<<<M / 16, 256, 0, stream>>>(A, w16, bias, C, M, K, Nc, act);
    };
    auto bnorm = [&](float* hbuf, const float* g, const float* b) {
        fill(bnsum, 0.f, 96);
        fill(bnsq, 0.f, 96);
        bn_stats<<<(N + 63) / 64, 96, 0, stream>>>(hbuf, bnsum, bnsq, N);
        bn_final<<<1, 96, 0, stream>>>(bnsum, bnsq, g, b, bnscl, bnsh, N);
        bn_apply_leaky<<<(N * HID + 255) / 256, 256, 0, stream>>>(hbuf, bnscl, bnsh, N * HID);
    };
    auto gcn = [&](const float* hin, const float* W, const float* g, const float* b,
                   float* hh, float* hout) {
        gemm(hin, W, nullptr, hh, N, HID, HID, 0);
        fill(hout, 0.f, N * HID);
        gcn_scatter<<<(EN * 32 + 255) / 256, 256, 0, stream>>>(ei, E, EN, dinv, hh, hout);
        bnorm(hout, g, b);
    };
    auto gat = [&](const float* hin, const float* W, const float* asr, const float* ads,
                   const float* g, const float* b, float* hh, float* hout) {
        gemm(hin, W, nullptr, hh, N, HID, HID, 0);
        gat_coef<<<(N * HEADS + 255) / 256, 256, 0, stream>>>(hh, asr, ads, as_, ad_, N);
        fillu(menc, 0u, N * HEADS);
        fill(ssum, 0.f, N * HEADS);
        gat_max<<<(EN * HEADS + 255) / 256, 256, 0, stream>>>(ei, E, EN, as_, ad_, menc);
        gat_expsum<<<(EN * HEADS + 255) / 256, 256, 0, stream>>>(ei, E, EN, as_, ad_, menc, alp, ssum);
        fill(hout, 0.f, N * HID);
        gat_scatter<<<(EN * 32 + 255) / 256, 256, 0, stream>>>(ei, E, EN, hh, alp, ssum, hout);
        bnorm(hout, g, b);
    };

    // 1) LayerNorm + projection
    ln_kernel<<<(N + 7) / 8, 256, 0, stream>>>(x, ln_g, ln_b, x_res, N);
    gemm(x_res, proj_W, proj_b, hb0, N, DIN, HID, 1);        // leaky

    // 2) degrees (once — same edge list for both GCN layers)
    fill(dinv, 1.0f, N);                                     // self-loop contributes 1
    deg_edges<<<(E + 255) / 256, 256, 0, stream>>>(ei, E, dinv);
    deg_fin<<<(N + 255) / 256, 256, 0, stream>>>(dinv, N);

    // 3) four conv layers (buffer rotation)
    gcn(hb0, W0, bng0, bnb0, hb1, hb2);                      // h -> hb2
    gat(hb2, W1, asrc1, adst1, bng1, bnb1, hb0, hb1);        // h -> hb1
    gcn(hb1, W2, bng2, bnb2, hb2, hb0);                      // h -> hb0
    gat(hb0, W3, asrc3, adst3, bng3, bnb3, hb1, hb2);        // h -> hb2 (final)

    // 4) GlobalAttention pooling
    gemm(hb2, gateW1, gateb1, hb0, N, HID, HID, 2);          // tanh -> gate1 in hb0
    gate_dot<<<(N * 32 + 255) / 256, 256, 0, stream>>>(hb0, gateW2, gate, N);
    fillu(m2, 0u, GNUM);
    fill(psum, 0.f, 2 * GNUM);                               // psum + cnt
    fill(gemb, 0.f, GNUM * HID);
    fill(xsum, 0.f, GNUM * DIN);
    pool_max<<<(N + 255) / 256, 256, 0, stream>>>(batch, gate, m2, N);
    pool_expsum<<<(N + 255) / 256, 256, 0, stream>>>(batch, gate, m2, psum, N);
    pool_scatter<<<(N * 32 + 255) / 256, 256, 0, stream>>>(batch, gate, psum, hb2, x_res,
                                                           gemb, xsum, cnt, N);

    // 5) readout: concat -> GEMM -> BatchNorm over graphs
    build_f<<<(GNUM * FDIM + 255) / 256, 256, 0, stream>>>(gemb, xsum, cnt, fbuf);
    gemm(fbuf, out_W, nullptr, obuf, GNUM, FDIM, OUTF, 0);
    final_bn<<<1, OUTF, 0, stream>>>(obuf, obn_g, obn_b, (float*)d_out);
}
